// EfficientMultiTensorProduct_34033320853904
// MI455X (gfx1250) — matrix operand order repaired and checked by reading, not compiled, not run on hardware
//
#include <hip/hip_runtime.h>

// ---------------------------------------------------------------------------
// EfficientMultiTensorProduct on gfx1250 (MI455X), monomial-basis WMMA form.
//
//   sh2_slice = PairMonomials  (NT x 136) x A2 (136 x 28)
//   sh3_slice = TripMonomials  (NT x 816) x A3 (816 x 28)
//   out       = w0*feat + w1*sh2 + w2*sh3   (gathered to packed irreps)
//
// A2/A3 are built on-device from Y/H2/H3, pre-swizzled into the wave32 WMMA
// B-fragment layout, DMA'd once per block into LDS by the Tensor Data Mover,
// and consumed by v_wmma_f32_16x16x32_bf16.
// ---------------------------------------------------------------------------

#define NN      1024
#define DD      128
#define EE      10
#define NT      (NN * DD)        // 131072 rows
#define NPAIR   136              // i<=j pairs of 16
#define NTRIP   816              // i<=j<=k triples of 16
#define K2P     160              // pairs padded to 5 k-steps
#define K3P     832              // triples padded to 26 k-steps
#define KTOT    (K2P + K3P)      // 992 = 31 k-steps
#define NSTEP   (KTOT / 32)      // 31
#define NSTEP2  (K2P / 32)       // 5
#define NOUTS   28               // 4 l-values x 7 m-values

#define CHUNK_COLS  512          // K processed in two chunks (16+15 steps)
#define MONO_STRIDE 520          // bf16 per LDS row (1040B: 16B-mult, bank-spread)
#define FEAT_STRIDE 20           // 16 feats + 1.0 + 0.0 + pad
#define WAVE_SMEM   (16*FEAT_STRIDE*4 + 16*12*4 + 16*MONO_STRIDE*2)  // 18688B
#define BFRAG_ELEMS (2 * NSTEP * 32 * 16)      // 31744 bf16
#define BFRAG_BYTES (BFRAG_ELEMS * 2)          // 63488B
#define BFRAG_LDS_OFF (4 * WAVE_SMEM)          // 74752
#define TOTAL_SMEM  (BFRAG_LDS_OFF + BFRAG_BYTES)  // 138240B -> 2 blocks/WGP

typedef __attribute__((ext_vector_type(16))) __bf16 v16bf;
typedef __attribute__((ext_vector_type(8)))  __bf16 v8bf;
typedef __attribute__((ext_vector_type(8)))  float  v8f;
typedef __attribute__((ext_vector_type(4)))  unsigned int u32x4;
typedef __attribute__((ext_vector_type(8)))  int i32x8;
typedef __attribute__((ext_vector_type(4)))  int i32x4;

// packed SH index q (0..15) -> (l, m) in the (4,7) centered-m grid
__device__ __constant__ int c_lq[16] = {0,1,1,1,2,2,2,2,2,3,3,3,3,3,3,3};
__device__ __constant__ int c_mq[16] = {3,2,3,4,1,2,3,4,5,0,1,2,3,4,5,6};

__device__ inline void unrank_pair(int p, int& i, int& j) {
  i = 0;
  while (p >= 16 - i) { p -= 16 - i; ++i; }
  j = i + p;
}

__device__ inline void unrank_trip(int t, int& i, int& j, int& k) {
  i = 0;
  for (;;) {
    int cnt = (16 - i) * (17 - i) / 2;
    if (t < cnt) break;
    t -= cnt; ++i;
  }
  j = i;
  while (t >= 16 - j) { t -= 16 - j; ++j; }
  k = j + t;
}

// ---------------------------------------------------------------------------
// Prep 1: column -> (i,j,k) monomial factor table.
// Pairs become triples via the 1.0 slot (index 16); pads hit the 0.0 slot (17).
// ---------------------------------------------------------------------------
__global__ void build_coltab(unsigned* __restrict__ colTab) {
  int c = blockIdx.x * blockDim.x + threadIdx.x;
  if (c >= KTOT) return;
  int i = 17, j = 17, k = 17;
  if (c < NPAIR) {
    unrank_pair(c, i, j);
    k = 16;                               // multiply by 1.0
  } else if (c >= K2P && c < K2P + NTRIP) {
    unrank_trip(c - K2P, i, j, k);
  }
  colTab[c] = (unsigned)i | ((unsigned)j << 8) | ((unsigned)k << 16);
}

// ---------------------------------------------------------------------------
// Prep 2: coefficient matrices A2/A3 written directly in the wave32 WMMA
// B-fragment layout:  Bfrag[t][s][lane][e], K = 32s + (lane>=16 ? 16:0) + e,
// Ncol = 16t + (lane&15).  One thread per fragment element (31744 total).
// ---------------------------------------------------------------------------
__global__ void build_bfrag(const float* __restrict__ Yre, const float* __restrict__ Yim,
                            const float* __restrict__ H2re, const float* __restrict__ H2im,
                            const float* __restrict__ H3re, const float* __restrict__ H3im,
                            __bf16* __restrict__ Bfrag) {
  int idx = blockIdx.x * blockDim.x + threadIdx.x;
  if (idx >= BFRAG_ELEMS) return;
  int e  = idx & 15;
  int L  = (idx >> 4) & 31;
  int sg = idx >> 9;                 // t*NSTEP + s
  int s  = sg % NSTEP;
  int t  = sg / NSTEP;
  int K    = 32 * s + ((L & 16) ? 16 : 0) + e;
  int Ncol = 16 * t + (L & 15);

  float val = 0.f;
  if (Ncol < NOUTS) {
    int l  = Ncol / 7;
    int mo = Ncol % 7;
    if (K < K2P) {
      if (K < NPAIR) {
        int i, j; unrank_pair(K, i, j);
        int li = c_lq[i], mi = c_mq[i], lj = c_lq[j], mj = c_mq[j];
        if (mi + mj == mo + 3) {
          float mult = (i == j) ? 1.f : 2.f;
          int mt = mo + 3;
          float acc = 0.f;
          for (int u1 = 0; u1 < 7; ++u1) {
            float y1r = Yre[li*49 + mi*7 + u1], y1i = Yim[li*49 + mi*7 + u1];
            for (int u2 = 0; u2 < 7; ++u2) {
              float y2r = Yre[lj*49 + mj*7 + u2], y2i = Yim[lj*49 + mj*7 + u2];
              float pr = y1r*y2r - y1i*y2i;
              float pi = y1r*y2i + y1i*y2r;
              int u = u1 + u2;
              float hr = H2re[l*169 + mt*13 + u], hi = H2im[l*169 + mt*13 + u];
              acc += pr*hr - pi*hi;
            }
          }
          val = mult * acc;
        }
      }
    } else {
      int tt = K - K2P;
      if (tt < NTRIP) {
        int i, j, k; unrank_trip(tt, i, j, k);
        int li = c_lq[i], mi = c_mq[i];
        int lj = c_lq[j], mj = c_mq[j];
        int lk = c_lq[k], mk = c_mq[k];
        if (mi + mj + mk == mo + 6) {
          float mult = (i == j && j == k) ? 1.f
                     : ((i == j || j == k || i == k) ? 3.f : 6.f);
          int mt = mo + 6;
          float acc = 0.f;
          for (int u1 = 0; u1 < 7; ++u1) {
            float y1r = Yre[li*49 + mi*7 + u1], y1i = Yim[li*49 + mi*7 + u1];
            for (int u2 = 0; u2 < 7; ++u2) {
              float y2r = Yre[lj*49 + mj*7 + u2], y2i = Yim[lj*49 + mj*7 + u2];
              float p12r = y1r*y2r - y1i*y2i;
              float p12i = y1r*y2i + y1i*y2r;
              for (int u3 = 0; u3 < 7; ++u3) {
                float y3r = Yre[lk*49 + mk*7 + u3], y3i = Yim[lk*49 + mk*7 + u3];
                float pr = p12r*y3r - p12i*y3i;
                float pi = p12r*y3i + p12i*y3r;
                int u = u1 + u2 + u3;
                float hr = H3re[l*361 + mt*19 + u], hi = H3im[l*361 + mt*19 + u];
                acc += pr*hr - pi*hi;
              }
            }
          }
          val = mult * acc;
        }
      }
    }
  }
  Bfrag[idx] = (__bf16)val;
}

// ---------------------------------------------------------------------------
// Main kernel: 1 wave = 1 tile of 16 (n,d) rows, 4 waves/block, 2048 blocks.
// B matrix TDM'd into LDS once per block; K processed in two 512-col chunks.
// ---------------------------------------------------------------------------
__global__ void __launch_bounds__(128)
emtp_main(const float* __restrict__ feat, const float* __restrict__ atype,
          const float* __restrict__ wts, const unsigned* __restrict__ colTab,
          const __bf16* __restrict__ Bfrag, float* __restrict__ out) {
  extern __shared__ char smem[];
  const int wave = threadIdx.x >> 5;
  const int lane = threadIdx.x & 31;
  char*   wbase = smem + wave * WAVE_SMEM;
  float*  featL = (float*)wbase;                                 // 16 x 20 f32
  float*  wL    = (float*)(wbase + 16 * FEAT_STRIDE * 4);        // 16 x 12 f32
  __bf16* mono  = (__bf16*)(wbase + 16*FEAT_STRIDE*4 + 16*12*4); // 16 x 520 bf16
  __bf16* bfragLds = (__bf16*)(smem + BFRAG_LDS_OFF);            // 63488 B

  const int tileBase = (blockIdx.x * 4 + wave) * 16;  // 128 % 16 == 0 -> n fixed
  const int n  = tileBase >> 7;
  const int d0 = tileBase & 127;

  // ---- Tensor Data Mover: stage the pre-swizzled coefficient fragments in
  // LDS once per block (7936 x 8B elements, 1-D tile, TENSORcnt-tracked).
  if (wave == 0) {
    unsigned long long ga = (unsigned long long)(size_t)Bfrag;
    unsigned ldsOff = (unsigned)(size_t)bfragLds;
    const unsigned nelem = BFRAG_BYTES / 8;            // 7936, fits tile_dim0
    u32x4 g0 = { 1u,                                   // count=1 (valid D#)
                 ldsOff,                               // lds_addr (bytes)
                 (unsigned)ga,                         // global_addr[31:0]
                 (unsigned)((ga >> 32) & 0x01FFFFFFu) | 0x80000000u }; // +type=2
    i32x8 g1 = { (int)(3u << 16),                      // data_size = 8B
                 (int)(nelem << 16),                   // tensor_dim0[15:0]
                 (int)((nelem >> 16) | (1u << 16)),    // td0[31:16] | td1[15:0]=1
                 (int)(nelem << 16),                   // td1 hi=0 | tile_dim0
                 0,                                    // tile_dim1=0, tile_dim2=0
                 (int)nelem,                           // tensor_dim0_stride lo32
                 0, 0 };
    i32x4 gz4 = { 0, 0, 0, 0 };
    i32x8 gz8 = { 0, 0, 0, 0, 0, 0, 0, 0 };
    __builtin_amdgcn_tensor_load_to_lds(g0, g1, gz4, gz4, gz8, 0);
  }

  // ---- phase 1: stage features (+1.0/+0.0 slots) and mixing weights w[k,l]
  {
    int row = lane >> 1, half = lane & 1;
    const float4* src = (const float4*)(feat + (size_t)(tileBase + row) * 16 + half * 8);
    float4 a = src[0], b = src[1];
    float* dst = featL + row * FEAT_STRIDE + half * 8;
    dst[0]=a.x; dst[1]=a.y; dst[2]=a.z; dst[3]=a.w;
    dst[4]=b.x; dst[5]=b.y; dst[6]=b.z; dst[7]=b.w;
    if (half) {
      float* ex = featL + row * FEAT_STRIDE;
      ex[16] = 1.f; ex[17] = 0.f; ex[18] = 0.f; ex[19] = 0.f;
    }
    for (int it = lane; it < 192; it += 32) {          // w[row][k][l], 16x3x4
      int row2 = it / 12, k = (it % 12) >> 2, l = it & 3;
      int dd = d0 + row2;
      float acc = 0.f;
      #pragma unroll
      for (int e2 = 0; e2 < EE; ++e2)
        acc += atype[n * EE + e2] * wts[((k * EE + e2) * DD + dd) * 4 + l];
      wL[it] = acc;
    }
  }
  // featL/wL/mono are intra-wave: same-wave LDS ordering fence is sufficient.
  asm volatile("s_wait_dscnt 0" ::: "memory");

  auto buildMono = [&](int c0, int c1) {
    for (int c = c0 + lane; c < c1; c += 32) {
      unsigned tab = colTab[c];
      int i = tab & 255, j = (tab >> 8) & 255, k = (tab >> 16) & 255;
      #pragma unroll
      for (int row = 0; row < 16; ++row) {
        const float* fr = featL + row * FEAT_STRIDE;
        mono[row * MONO_STRIDE + (c - c0)] = (__bf16)(fr[i] * fr[j] * fr[k]);
      }
    }
  };
  auto loadA = [&](int s, int cbase) -> v16bf {
    const int mrow = lane & 15;
    const int koff = (lane & 16) ? 16 : 0;
    const __bf16* ap = mono + mrow * MONO_STRIDE + (32 * s - cbase) + koff;
    v8bf lo = *(const v8bf*)ap;            // K = 32s+koff    .. +7
    v8bf hi = *(const v8bf*)(ap + 8);      // K = 32s+koff+8  .. +15
    v16bf A;
    #pragma unroll
    for (int e = 0; e < 8; ++e) { A[e] = lo[e]; A[e + 8] = hi[e]; }
    return A;
  };
  auto loadB = [&](int t, int s) -> v16bf {
    const __bf16* bp = bfragLds + ((t * NSTEP + s) * 32 + lane) * 16;
    v8bf lo = *(const v8bf*)bp;
    v8bf hi = *(const v8bf*)(bp + 8);
    v16bf B;
    #pragma unroll
    for (int e = 0; e < 8; ++e) { B[e] = lo[e]; B[e + 8] = hi[e]; }
    return B;
  };

  v8f c2a = {0,0,0,0,0,0,0,0}, c2b = {0,0,0,0,0,0,0,0};
  v8f c3a = {0,0,0,0,0,0,0,0}, c3b = {0,0,0,0,0,0,0,0};

  // ---- chunk 0: monomial cols [0,512) = k-steps 0..15
  buildMono(0, CHUNK_COLS);
  if (wave == 0) __builtin_amdgcn_s_wait_tensorcnt(0);  // TDM done
  __syncthreads();                                      // publish bfragLds
  #pragma unroll
  for (int s = 0; s < NSTEP2; ++s) {        // pair-monomial steps -> sh2
    v16bf A = loadA(s, 0);
    c2a = __builtin_amdgcn_wmma_f32_16x16x32_bf16(false, A, false, loadB(0, s),
                                                  (short)0, c2a, false, false);
    c2b = __builtin_amdgcn_wmma_f32_16x16x32_bf16(false, A, false, loadB(1, s),
                                                  (short)0, c2b, false, false);
  }
  #pragma unroll
  for (int s = NSTEP2; s < 16; ++s) {       // triple-monomial steps -> sh3
    v16bf A = loadA(s, 0);
    c3a = __builtin_amdgcn_wmma_f32_16x16x32_bf16(false, A, false, loadB(0, s),
                                                  (short)0, c3a, false, false);
    c3b = __builtin_amdgcn_wmma_f32_16x16x32_bf16(false, A, false, loadB(1, s),
                                                  (short)0, c3b, false, false);
  }
  asm volatile("s_wait_dscnt 0" ::: "memory");   // WAR: A reads before overwrite

  // ---- chunk 1: monomial cols [512,992) = k-steps 16..30
  buildMono(CHUNK_COLS, KTOT);
  asm volatile("s_wait_dscnt 0" ::: "memory");
  #pragma unroll
  for (int s = 16; s < NSTEP; ++s) {
    v16bf A = loadA(s, CHUNK_COLS);
    c3a = __builtin_amdgcn_wmma_f32_16x16x32_bf16(false, A, false, loadB(0, s),
                                                  (short)0, c3a, false, false);
    c3b = __builtin_amdgcn_wmma_f32_16x16x32_bf16(false, A, false, loadB(1, s),
                                                  (short)0, c3b, false, false);
  }

  // ---- epilogue: combine with per-l weights, gather to packed irreps output.
  // f32 C layout: VGPR v holds row M = v + (lane>=16 ? 8:0), column N = lane&15.
  const int ncol  = lane & 15;
  const int rbase = (lane & 16) ? 8 : 0;
  #pragma unroll
  for (int t = 0; t < 2; ++t) {
    int Ncol = ncol + 16 * t;
    if (Ncol >= NOUTS) continue;
    int l = Ncol / 7, m = Ncol % 7;
    int dm = m - 3; if (dm < 0) dm = -dm;
    if (dm > l) continue;                   // only mask positions exist in output
    int q      = l * l + m - (3 - l);       // packed SH index
    int base_l = DD * l * l;                // 0,128,512,1152
    const v8f& A2 = t ? c2b : c2a;
    const v8f& A3 = t ? c3b : c3a;
    #pragma unroll
    for (int v = 0; v < 8; ++v) {
      int row = rbase + v;
      int dd  = d0 + row;
      float w0 = wL[row * 12 + 0 + l];
      float w1 = wL[row * 12 + 4 + l];
      float w2 = wL[row * 12 + 8 + l];
      float f1 = featL[row * FEAT_STRIDE + q];
      float valr = w0 * f1 + w1 * A2[v] + w2 * A3[v];
      out[(size_t)n * 2048 + base_l + dd * (2 * l + 1) + (m - (3 - l))] = valr;
    }
  }
}

// ---------------------------------------------------------------------------
extern "C" void kernel_launch(void* const* d_in, const int* in_sizes, int n_in,
                              void* d_out, int out_size, void* d_ws, size_t ws_size,
                              hipStream_t stream) {
  (void)in_sizes; (void)n_in; (void)out_size; (void)ws_size;
  const float* feat  = (const float*)d_in[0];
  const float* atype = (const float*)d_in[1];
  const float* wts   = (const float*)d_in[2];
  const float* Yre   = (const float*)d_in[3];
  const float* Yim   = (const float*)d_in[4];
  const float* H2re  = (const float*)d_in[5];
  const float* H2im  = (const float*)d_in[6];
  const float* H3re  = (const float*)d_in[7];
  const float* H3im  = (const float*)d_in[8];

  unsigned* colTab = (unsigned*)d_ws;                       // 3968 B (rsvd 4096)
  __bf16*   Bfrag  = (__bf16*)((char*)d_ws + 4096);         // 63488 B
  float*    out    = (float*)d_out;

  build_coltab<<<1, 1024, 0, stream>>>(colTab);
  build_bfrag<<<124, 256, 0, stream>>>(Yre, Yim, H2re, H2im, H3re, H3im, Bfrag);
  emtp_main<<<NT / 64, 128, TOTAL_SMEM, stream>>>(feat, atype, wts, colTab,
                                                  Bfrag, out);
}
